// Llama3_68135361184133
// MI455X (gfx1250) — compile-verified, matrix operand order checked by reading, not run on hardware
//
#include <hip/hip_runtime.h>
#include <hip/hip_bf16.h>

// ---------------------------------------------------------------------------
// PatchTST + single Llama3 decoder layer forward for MI455X (gfx1250).
// Big GEMMs: bf16 v_wmma_f32_16x16x32_bf16, 128x128 block tile, BK=32,
// double-buffered LDS staged via GLOBAL_LOAD_ASYNC_TO_LDS_B128 (ASYNCcnt).
// Weights converted fp32->bf16 into workspace each launch (deterministic).
// Workspace layout requires ~1.65 GB (harness-provided d_ws).
// ---------------------------------------------------------------------------

typedef __attribute__((ext_vector_type(16))) __bf16 v16bf;
typedef __attribute__((ext_vector_type(8)))  float  v8f;
typedef unsigned short u16;
typedef unsigned int   u32;

static constexpr int Bc = 16, Lc = 512, Mc = 7;
static constexpr int Nc = 64, Pc = 16, Sc = 8;
static constexpr int Dc = 4096, Hc = 32, KVc = 8, HDc = 128, DFFc = 14336;
static constexpr int FCc = 128, PREDc = 96;
static constexpr int BMc = Bc * Mc;          // 112 sequences
static constexpr int TOK = BMc * Nc;         // 7168 tokens
static constexpr float EPSc = 1e-5f;

__device__ __forceinline__ u16 f32_bf16(float f) {
    u32 u = __float_as_uint(f);
    u = (u + 0x7FFFu + ((u >> 16) & 1u)) >> 16;   // round-to-nearest-even
    return (u16)u;
}

// ---------------------------------------------------------------------------
// CDNA5 async global->LDS copy (ASYNCcnt) with portable fallback
// ---------------------------------------------------------------------------
#if defined(__gfx1250__) && __has_builtin(__builtin_amdgcn_global_load_async_to_lds_b128)
#define USE_ASYNC_LDS 1
#else
#define USE_ASYNC_LDS 0
#endif

typedef int v4i __attribute__((vector_size(16)));
typedef __attribute__((address_space(1))) v4i gv4i_t;   // global (AS1)
typedef __attribute__((address_space(3))) v4i lv4i_t;   // LDS (AS3)

__device__ __forceinline__ void async_copy16(const u16* __restrict__ g, u16* l) {
#if USE_ASYNC_LDS
    __builtin_amdgcn_global_load_async_to_lds_b128((gv4i_t*)g, (lv4i_t*)l, 0, 0);
#else
    *reinterpret_cast<uint4*>(l) = *reinterpret_cast<const uint4*>(g);
#endif
}

template <int N>
__device__ __forceinline__ void wait_async() {
#if defined(__gfx1250__) && __has_builtin(__builtin_amdgcn_s_wait_asynccnt)
    __builtin_amdgcn_s_wait_asynccnt(N);
#endif
}

// ---------------------------------------------------------------------------
// elementwise f32 -> bf16 cast (grid-stride)
// ---------------------------------------------------------------------------
__global__ void cast_bf16_kernel(const float* __restrict__ src,
                                 u16* __restrict__ dst, size_t n) {
    size_t i = (size_t)blockIdx.x * blockDim.x + threadIdx.x;
    size_t stride = (size_t)gridDim.x * blockDim.x;
    for (; i < n; i += stride) dst[i] = f32_bf16(src[i]);
}

__global__ void zero_kernel(float* __restrict__ p, int n) {
    int i = blockIdx.x * blockDim.x + threadIdx.x;
    if (i < n) p[i] = 0.f;
}

// ---------------------------------------------------------------------------
// RevIN instance-norm stats per (b, m) channel over L
// ---------------------------------------------------------------------------
__global__ __launch_bounds__(128)
void stats_kernel(const float* __restrict__ x, float* __restrict__ mean,
                  float* __restrict__ stdev) {
    int bm = blockIdx.x;
    int b = bm / Mc, m = bm % Mc;
    const float* xb = x + (size_t)b * Lc * Mc + m;
    float s = 0.f, s2 = 0.f;
    for (int l = threadIdx.x; l < Lc; l += blockDim.x) {
        float v = xb[(size_t)l * Mc];
        s += v; s2 += v * v;
    }
    __shared__ float smA[4], smB[4];
    for (int o = 16; o > 0; o >>= 1) { s += __shfl_down(s, o, 32); s2 += __shfl_down(s2, o, 32); }
    int lane = threadIdx.x & 31, w = threadIdx.x >> 5;
    if (lane == 0) { smA[w] = s; smB[w] = s2; }
    __syncthreads();
    if (threadIdx.x == 0) {
        float ts = 0.f, ts2 = 0.f;
        for (int i = 0; i < 4; ++i) { ts += smA[i]; ts2 += smB[i]; }
        float mu = ts / Lc;
        float var = ts2 / Lc - mu * mu;
        mean[bm] = mu;
        stdev[bm] = sqrtf(var + EPSc);
    }
}

// ---------------------------------------------------------------------------
// patch unfold (replication pad) + embed: h = patches @ W_in^T + b_in  (K=16)
// ---------------------------------------------------------------------------
__global__ __launch_bounds__(256)
void patch_embed_kernel(const float* __restrict__ x, const float* __restrict__ W_in,
                        const float* __restrict__ b_in, const float* __restrict__ mean,
                        const float* __restrict__ stdev, float* __restrict__ h) {
    int token = blockIdx.x;
    int bm = token >> 6, n = token & 63;
    int b = bm / Mc, m = bm % Mc;
    __shared__ float patch[Pc];
    if (threadIdx.x < Pc) {
        int l = n * Sc + threadIdx.x;
        if (l > Lc - 1) l = Lc - 1;                 // replication pad
        float v = x[(size_t)b * Lc * Mc + (size_t)l * Mc + m];
        patch[threadIdx.x] = (v - mean[bm]) / stdev[bm];
    }
    __syncthreads();
    float* hr = h + (size_t)token * Dc;
    for (int d = threadIdx.x; d < Dc; d += 256) {
        const float* wr = W_in + (size_t)d * Pc;
        float acc = b_in[d];
#pragma unroll
        for (int p = 0; p < Pc; ++p) acc += patch[p] * wr[p];
        hr[d] = acc;
    }
}

// ---------------------------------------------------------------------------
// RMSNorm row (D=4096) -> bf16
// ---------------------------------------------------------------------------
__global__ __launch_bounds__(256)
void rmsnorm_bf16_kernel(const float* __restrict__ h, const float* __restrict__ w,
                         u16* __restrict__ out) {
    int row = blockIdx.x;
    const float* hr = h + (size_t)row * Dc;
    float s2 = 0.f;
    for (int d = threadIdx.x; d < Dc; d += 256) { float v = hr[d]; s2 += v * v; }
    __shared__ float sm[8];
    __shared__ float stot;
    for (int o = 16; o > 0; o >>= 1) s2 += __shfl_down(s2, o, 32);
    if ((threadIdx.x & 31) == 0) sm[threadIdx.x >> 5] = s2;
    __syncthreads();
    if (threadIdx.x == 0) {
        float t = 0.f;
        for (int i = 0; i < 8; ++i) t += sm[i];
        stot = rsqrtf(t / (float)Dc + EPSc);
    }
    __syncthreads();
    float inv = stot;
    u16* orow = out + (size_t)row * Dc;
    for (int d = threadIdx.x; d < Dc; d += 256) orow[d] = f32_bf16(hr[d] * inv * w[d]);
}

// ---------------------------------------------------------------------------
// WMMA bf16 GEMM: C[M,N] = A[M,K] * W[N,K]^T  (+ fused epilogues)
//  block: 256 threads = 8 waves (2x4), tile 128x128, BK=32, double-buffered
//  async LDS staging; per-wave 64x32 via 4x2 grid of v_wmma_f32_16x16x32_bf16.
//  REQUIRES: M, N multiples of 128, K multiple of 32 (true for all callers).
// ---------------------------------------------------------------------------
#define LDSS 40   // padded LDS row stride in bf16 elements (80B, 16B-aligned)

union FragBF { uint4 q[2]; v16bf v; };

__device__ __forceinline__ v16bf lds_frag(const u16* s, int row, int lane) {
    // CDNA5 16-bit operand layout: lanes 0-15 hold K {0..7,16..23},
    // lanes 16-31 hold K {8..15,24..31} for the same row.
    int kbase = (lane >> 4) << 3;
    const u16* p = s + row * LDSS;
    FragBF f;
    f.q[0] = *reinterpret_cast<const uint4*>(p + kbase);
    f.q[1] = *reinterpret_cast<const uint4*>(p + kbase + 16);
    return f.v;
}

__global__ __launch_bounds__(256)
void gemm_bf16_wmma(const u16* __restrict__ A, const u16* __restrict__ Bw,
                    void* __restrict__ Cout, int Mrows, int Nout, int K,
                    const float* __restrict__ residual,
                    const float* __restrict__ gate, int outBf16) {
    __shared__ u16 As[2][128 * LDSS];
    __shared__ u16 Bs[2][128 * LDSS];
    const int tid = threadIdx.x;
    const int lane = tid & 31;
    const int wave = tid >> 5;
    const int waveM = wave >> 2;     // 0..1
    const int waveN = wave & 3;      // 0..3
    const int blockN0 = blockIdx.x * 128;
    const int blockM0 = blockIdx.y * 128;

    // this thread's staging slots: rows r0 and r0+64, 8 bf16 (16B) at col c0
    const int r0 = tid >> 2;
    const int c0 = (tid & 3) * 8;
    const u16* aG = A + (size_t)(blockM0 + r0) * K + c0;
    const u16* bG = Bw + (size_t)(blockN0 + r0) * K + c0;
    const size_t half = (size_t)64 * K;

    v8f acc[4][2];
#pragma unroll
    for (int mi = 0; mi < 4; ++mi)
#pragma unroll
        for (int ni = 0; ni < 2; ++ni) acc[mi][ni] = (v8f){};

    auto stage = [&](int buf, int k0) {
        async_copy16(aG + k0,        &As[buf][r0 * LDSS + c0]);
        async_copy16(aG + half + k0, &As[buf][(r0 + 64) * LDSS + c0]);
        async_copy16(bG + k0,        &Bs[buf][r0 * LDSS + c0]);
        async_copy16(bG + half + k0, &Bs[buf][(r0 + 64) * LDSS + c0]);
    };

    const int nt = K >> 5;
    stage(0, 0);
    for (int t = 0; t < nt; ++t) {
        const int buf = t & 1;
        if (t + 1 < nt) {
            stage(buf ^ 1, (t + 1) << 5);   // prefetch next tile (4 async ops)
            wait_async<4>();                // current tile complete
        } else {
            wait_async<0>();
        }
        __syncthreads();
        v16bf bfrag[2];
#pragma unroll
        for (int ni = 0; ni < 2; ++ni)
            bfrag[ni] = lds_frag(Bs[buf], waveN * 32 + ni * 16 + (lane & 15), lane);
#pragma unroll
        for (int mi = 0; mi < 4; ++mi) {
            v16bf afrag = lds_frag(As[buf], waveM * 64 + mi * 16 + (lane & 15), lane);
#pragma unroll
            for (int ni = 0; ni < 2; ++ni)
                acc[mi][ni] = __builtin_amdgcn_wmma_f32_16x16x32_bf16(
                    false, afrag, false, bfrag[ni], (short)0, acc[mi][ni], false, false);
        }
        __syncthreads();
    }

    // epilogue: C layout VGPR i -> (M = base+i | base+8+i, N = lane&15)
#pragma unroll
    for (int mi = 0; mi < 4; ++mi)
#pragma unroll
        for (int ni = 0; ni < 2; ++ni) {
            int colG = blockN0 + waveN * 32 + ni * 16 + (lane & 15);
            int rowBase = blockM0 + waveM * 64 + mi * 16 + ((lane >> 4) << 3);
#pragma unroll
            for (int i = 0; i < 8; ++i) {
                int rowG = rowBase + i;
                float v = acc[mi][ni][i];
                size_t off = (size_t)rowG * Nout + colG;
                if (gate) { float g = gate[off]; v *= g / (1.f + __expf(-g)); } // silu(g)*v
                if (residual) v += residual[off];
                if (outBf16) ((u16*)Cout)[off] = f32_bf16(v);
                else         ((float*)Cout)[off] = v;
            }
        }
    (void)Mrows;
}

// split-K variant for the FC head (M=112, N=128, K=262144): atomic f32 accum,
// synchronous staging with M-edge zero fill.
__global__ __launch_bounds__(256)
void gemm_bf16_wmma_splitk(const u16* __restrict__ A, const u16* __restrict__ Bw,
                           float* __restrict__ Cout, int Mrows, int Nout, int K,
                           int kchunk) {
    __shared__ u16 As[128 * LDSS];
    __shared__ u16 Bs[128 * LDSS];
    const int tid = threadIdx.x;
    const int lane = tid & 31;
    const int wave = tid >> 5;
    const int waveM = wave >> 2;
    const int waveN = wave & 3;
    const int blockN0 = blockIdx.x * 128;
    const int blockM0 = blockIdx.y * 128;
    const int kstart = blockIdx.z * kchunk;
    int kend = kstart + kchunk; if (kend > K) kend = K;

    v8f acc[4][2];
#pragma unroll
    for (int mi = 0; mi < 4; ++mi)
#pragma unroll
        for (int ni = 0; ni < 2; ++ni) acc[mi][ni] = (v8f){};

    for (int k0 = kstart; k0 < kend; k0 += 32) {
#pragma unroll
        for (int p = 0; p < 2; ++p) {
            int idx = p * 2048 + tid * 8;
            int row = idx >> 5, col = idx & 31;
            uint4 av = make_uint4(0u, 0u, 0u, 0u);
            int ar = blockM0 + row;
            if (ar < Mrows)
                av = *reinterpret_cast<const uint4*>(A + (size_t)ar * K + k0 + col);
            *reinterpret_cast<uint4*>(&As[row * LDSS + col]) = av;
            uint4 bv = make_uint4(0u, 0u, 0u, 0u);
            int br = blockN0 + row;
            if (br < Nout)
                bv = *reinterpret_cast<const uint4*>(Bw + (size_t)br * K + k0 + col);
            *reinterpret_cast<uint4*>(&Bs[row * LDSS + col]) = bv;
        }
        __syncthreads();
        v16bf bfrag[2];
#pragma unroll
        for (int ni = 0; ni < 2; ++ni)
            bfrag[ni] = lds_frag(Bs, waveN * 32 + ni * 16 + (lane & 15), lane);
#pragma unroll
        for (int mi = 0; mi < 4; ++mi) {
            v16bf afrag = lds_frag(As, waveM * 64 + mi * 16 + (lane & 15), lane);
#pragma unroll
            for (int ni = 0; ni < 2; ++ni)
                acc[mi][ni] = __builtin_amdgcn_wmma_f32_16x16x32_bf16(
                    false, afrag, false, bfrag[ni], (short)0, acc[mi][ni], false, false);
        }
        __syncthreads();
    }
#pragma unroll
    for (int mi = 0; mi < 4; ++mi)
#pragma unroll
        for (int ni = 0; ni < 2; ++ni) {
            int colG = blockN0 + waveN * 32 + ni * 16 + (lane & 15);
            int rowBase = blockM0 + waveM * 64 + mi * 16 + ((lane >> 4) << 3);
#pragma unroll
            for (int i = 0; i < 8; ++i) {
                int rowG = rowBase + i;
                if (rowG >= Mrows || colG >= Nout) continue;
                atomicAdd(&Cout[(size_t)rowG * Nout + colG], acc[mi][ni][i]);
            }
        }
}

// ---------------------------------------------------------------------------
// RoPE in-place on q (32 heads) and k (8 heads), HD=128, theta=500000
// ---------------------------------------------------------------------------
__global__ __launch_bounds__(256)
void rope_kernel(float* __restrict__ q, float* __restrict__ k) {
    int token = blockIdx.x;
    float pos = (float)(token & (Nc - 1));
    const float lnTheta = 13.12236338f;               // ln(500000)
    for (int t = threadIdx.x; t < (Hc + KVc) * 64; t += 256) {
        float* base; int j;
        if (t < Hc * 64) {
            int hh = t >> 6; j = t & 63;
            base = q + (size_t)token * (Hc * HDc) + hh * HDc;
        } else {
            int t2 = t - Hc * 64; int hh = t2 >> 6; j = t2 & 63;
            base = k + (size_t)token * (KVc * HDc) + hh * HDc;
        }
        float freq = __expf(-((float)(2 * j) * (1.0f / HDc)) * lnTheta);
        float ang = pos * freq;
        float c = __cosf(ang), s = __sinf(ang);
        float x1 = base[j], x2 = base[j + 64];
        base[j]      = x1 * c - x2 * s;
        base[j + 64] = x2 * c + x1 * s;
    }
}

// ---------------------------------------------------------------------------
// causal GQA attention over N=64 per (bm, head); K staged in LDS
// ---------------------------------------------------------------------------
__global__ __launch_bounds__(64)
void attn_kernel(const float* __restrict__ q, const float* __restrict__ k,
                 const float* __restrict__ v, float* __restrict__ o) {
    int hh = blockIdx.x;         // 0..31
    int bm = blockIdx.y;         // 0..111
    int kh = hh >> 2;            // H / KV = 4
    __shared__ float Ks[Nc * HDc];       // 32 KB
    __shared__ float Sm[Nc * 65];        // ~16.6 KB
    int tid = threadIdx.x;
    for (int idx = tid; idx < Nc * 32; idx += 64) {    // 2048 float4
        int j = idx >> 5, c4 = idx & 31;
        const float4* src = reinterpret_cast<const float4*>(
            k + (size_t)(bm * Nc + j) * (KVc * HDc) + kh * HDc);
        reinterpret_cast<float4*>(&Ks[j * HDc])[c4] = src[c4];
    }
    __syncthreads();
    int i = tid;                                      // query row
    float* srow = &Sm[i * 65];
    for (int j = 0; j <= i; ++j) srow[j] = 0.f;
    const float* qrow = q + (size_t)(bm * Nc + i) * (Hc * HDc) + hh * HDc;
    for (int dc = 0; dc < 4; ++dc) {
        float4 qv[8];
        const float4* q4 = reinterpret_cast<const float4*>(qrow + dc * 32);
#pragma unroll
        for (int u = 0; u < 8; ++u) qv[u] = q4[u];
        for (int j = 0; j <= i; ++j) {
            const float4* k4 = reinterpret_cast<const float4*>(&Ks[j * HDc + dc * 32]);
            float p = 0.f;
#pragma unroll
            for (int u = 0; u < 8; ++u) {
                float4 kv = k4[u];
                p += qv[u].x * kv.x + qv[u].y * kv.y + qv[u].z * kv.z + qv[u].w * kv.w;
            }
            srow[j] += p;
        }
    }
    const float scale = 0.088388347648318447f;        // 1/sqrt(128)
    float mx = -1e30f;
    for (int j = 0; j <= i; ++j) { srow[j] *= scale; mx = fmaxf(mx, srow[j]); }
    float sum = 0.f;
    for (int j = 0; j <= i; ++j) { float e = __expf(srow[j] - mx); srow[j] = e; sum += e; }
    float inv = 1.f / sum;
    float* orow = o + (size_t)(bm * Nc + i) * (Hc * HDc) + hh * HDc;
    for (int dc = 0; dc < 4; ++dc) {
        float acc[32];
#pragma unroll
        for (int d = 0; d < 32; ++d) acc[d] = 0.f;
        for (int j = 0; j <= i; ++j) {
            float p = srow[j];
            const float* vr = v + (size_t)(bm * Nc + j) * (KVc * HDc) + kh * HDc + dc * 32;
#pragma unroll
            for (int d = 0; d < 32; ++d) acc[d] += p * vr[d];
        }
#pragma unroll
        for (int d = 0; d < 32; ++d) orow[dc * 32 + d] = acc[d] * inv;
    }
}

// ---------------------------------------------------------------------------
// head: leaky(z + b_fc) @ W_out^T + b_out, then RevIN denorm + transpose
// ---------------------------------------------------------------------------
__global__ __launch_bounds__(128)
void head_kernel(const float* __restrict__ z, const float* __restrict__ b_fc,
                 const float* __restrict__ W_out, const float* __restrict__ b_out,
                 const float* __restrict__ mean, const float* __restrict__ stdev,
                 float* __restrict__ out) {
    int bm = blockIdx.x, t = threadIdx.x;
    __shared__ float zs[FCc];
    float zv = z[(size_t)bm * FCc + t] + b_fc[t];
    zs[t] = zv > 0.f ? zv : 0.01f * zv;
    __syncthreads();
    if (t < PREDc) {
        float acc = b_out[t];
        const float* wr = W_out + (size_t)t * FCc;
#pragma unroll 4
        for (int c = 0; c < FCc; ++c) acc += zs[c] * wr[c];
        int b = bm / Mc, m = bm % Mc;
        out[(size_t)b * PREDc * Mc + (size_t)t * Mc + m] = acc * stdev[bm] + mean[bm];
    }
}

// ---------------------------------------------------------------------------
// launch
// ---------------------------------------------------------------------------
static inline size_t alup(size_t x) { return (x + 255) & ~(size_t)255; }

extern "C" void kernel_launch(void* const* d_in, const int* in_sizes, int n_in,
                              void* d_out, int out_size, void* d_ws, size_t ws_size,
                              hipStream_t stream) {
    const float* x        = (const float*)d_in[0];
    const float* W_in     = (const float*)d_in[1];
    const float* b_in     = (const float*)d_in[2];
    const float* attn_nw  = (const float*)d_in[3];
    const float* Wq       = (const float*)d_in[4];
    const float* Wk       = (const float*)d_in[5];
    const float* Wv       = (const float*)d_in[6];
    const float* Wo       = (const float*)d_in[7];
    const float* mlp_nw   = (const float*)d_in[8];
    const float* Wg       = (const float*)d_in[9];
    const float* Wu       = (const float*)d_in[10];
    const float* Wd       = (const float*)d_in[11];
    const float* final_nw = (const float*)d_in[12];
    const float* W_fc     = (const float*)d_in[13];
    const float* b_fc     = (const float*)d_in[14];
    const float* W_out    = (const float*)d_in[15];
    const float* b_out    = (const float*)d_in[16];
    float* out = (float*)d_out;

    // ---- workspace map (≈1.65 GB) ----
    char* base = (char*)d_ws;
    size_t cur = 0;
    auto grab = [&](size_t bytes) { size_t o = cur; cur = alup(cur + bytes); return o; };
    size_t o_stats = grab(2 * BMc * sizeof(float));                 // mean | stdev
    size_t o_h     = grab((size_t)TOK * Dc * 4);                    // h f32
    size_t o_hn    = grab((size_t)TOK * Dc * 2);                    // normed bf16
    size_t o_q     = grab((size_t)TOK * Hc * HDc * 4);
    size_t o_k     = grab((size_t)TOK * KVc * HDc * 4);
    size_t o_v     = grab((size_t)TOK * KVc * HDc * 4);
    size_t o_att   = grab((size_t)TOK * Dc * 4);                    // attn out f32
    size_t o_abf   = grab((size_t)TOK * Dc * 2);                    // attn out bf16
    size_t o_g     = grab((size_t)TOK * DFFc * 4);                  // gate f32
    size_t o_act   = grab((size_t)TOK * DFFc * 2);                  // silu(g)*u bf16
    size_t o_z     = grab((size_t)BMc * FCc * 4);
    size_t o_wq    = grab((size_t)Hc * HDc * Dc * 2);
    size_t o_wk    = grab((size_t)KVc * HDc * Dc * 2);
    size_t o_wv    = grab((size_t)KVc * HDc * Dc * 2);
    size_t o_wo    = grab((size_t)Dc * Hc * HDc * 2);
    size_t o_wg    = grab((size_t)DFFc * Dc * 2);
    size_t o_wu    = grab((size_t)DFFc * Dc * 2);
    size_t o_wd    = grab((size_t)Dc * DFFc * 2);
    size_t o_wfc   = grab((size_t)FCc * Nc * Dc * 2);
    (void)ws_size; (void)n_in; (void)in_sizes; (void)out_size;

    float* mean  = (float*)(base + o_stats);
    float* stdev = mean + BMc;
    float* hbuf  = (float*)(base + o_h);
    u16*   hn    = (u16*)(base + o_hn);
    float* qb    = (float*)(base + o_q);
    float* kb    = (float*)(base + o_k);
    float* vb    = (float*)(base + o_v);
    float* attb  = (float*)(base + o_att);
    u16*   abf   = (u16*)(base + o_abf);
    float* gbuf  = (float*)(base + o_g);
    u16*   act   = (u16*)(base + o_act);
    float* zbuf  = (float*)(base + o_z);
    u16 *wqb=(u16*)(base+o_wq), *wkb=(u16*)(base+o_wk), *wvb=(u16*)(base+o_wv);
    u16 *wob=(u16*)(base+o_wo), *wgb=(u16*)(base+o_wg), *wub=(u16*)(base+o_wu);
    u16 *wdb=(u16*)(base+o_wd), *wfcb=(u16*)(base+o_wfc);

    // ---- weight fp32 -> bf16 ----
    auto cast = [&](const float* s, u16* d, size_t n) {
        cast_bf16_kernel<<<4096, 256, 0, stream>>>(s, d, n);
    };
    cast(Wq,  wqb,  (size_t)Hc * HDc * Dc);
    cast(Wk,  wkb,  (size_t)KVc * HDc * Dc);
    cast(Wv,  wvb,  (size_t)KVc * HDc * Dc);
    cast(Wo,  wob,  (size_t)Dc * Hc * HDc);
    cast(Wg,  wgb,  (size_t)DFFc * Dc);
    cast(Wu,  wub,  (size_t)DFFc * Dc);
    cast(Wd,  wdb,  (size_t)Dc * DFFc);
    cast(W_fc, wfcb, (size_t)FCc * Nc * Dc);

    // ---- RevIN stats + patch embed ----
    stats_kernel<<<BMc, 128, 0, stream>>>(x, mean, stdev);
    patch_embed_kernel<<<TOK, 256, 0, stream>>>(x, W_in, b_in, mean, stdev, hbuf);

    dim3 blk(256);
    const int MT = TOK / 128;                      // 56 row tiles
    // ---- attention block ----
    rmsnorm_bf16_kernel<<<TOK, 256, 0, stream>>>(hbuf, attn_nw, hn);
    gemm_bf16_wmma<<<dim3(Dc / 128, MT), blk, 0, stream>>>(hn, wqb, qb, TOK, Dc, Dc, nullptr, nullptr, 0);
    gemm_bf16_wmma<<<dim3(KVc * HDc / 128, MT), blk, 0, stream>>>(hn, wkb, kb, TOK, KVc * HDc, Dc, nullptr, nullptr, 0);
    gemm_bf16_wmma<<<dim3(KVc * HDc / 128, MT), blk, 0, stream>>>(hn, wvb, vb, TOK, KVc * HDc, Dc, nullptr, nullptr, 0);
    rope_kernel<<<TOK, 256, 0, stream>>>(qb, kb);
    attn_kernel<<<dim3(Hc, BMc), 64, 0, stream>>>(qb, kb, vb, attb);
    cast_bf16_kernel<<<4096, 256, 0, stream>>>(attb, abf, (size_t)TOK * Dc);
    // h = h + o @ Wo^T   (residual fused, in-place)
    gemm_bf16_wmma<<<dim3(Dc / 128, MT), blk, 0, stream>>>(abf, wob, hbuf, TOK, Dc, Dc, hbuf, nullptr, 0);

    // ---- SwiGLU MLP ----
    rmsnorm_bf16_kernel<<<TOK, 256, 0, stream>>>(hbuf, mlp_nw, hn);
    gemm_bf16_wmma<<<dim3(DFFc / 128, MT), blk, 0, stream>>>(hn, wgb, gbuf, TOK, DFFc, Dc, nullptr, nullptr, 0);
    // act = bf16( silu(g) * (hn @ Wu^T) )   (gate fused)
    gemm_bf16_wmma<<<dim3(DFFc / 128, MT), blk, 0, stream>>>(hn, wub, act, TOK, DFFc, Dc, nullptr, gbuf, 1);
    // h = h + act @ Wd^T
    gemm_bf16_wmma<<<dim3(Dc / 128, MT), blk, 0, stream>>>(act, wdb, hbuf, TOK, Dc, DFFc, hbuf, nullptr, 0);

    // ---- final norm + FC head (split-K WMMA) ----
    rmsnorm_bf16_kernel<<<TOK, 256, 0, stream>>>(hbuf, final_nw, hn);
    zero_kernel<<<(BMc * FCc + 255) / 256, 256, 0, stream>>>(zbuf, BMc * FCc);
    const int Kfc = Nc * Dc;                       // 262144
    const int ksplit = 64, kchunk = Kfc / ksplit;  // 4096 per block
    gemm_bf16_wmma_splitk<<<dim3(1, 1, ksplit), blk, 0, stream>>>(
        hn, wfcb, zbuf, BMc, FCc, Kfc, kchunk);
    head_kernel<<<BMc, 128, 0, stream>>>(zbuf, b_fc, W_out, b_out, mean, stdev, out);
}
